// Encoder_13116830122073
// MI455X (gfx1250) — compile-verified, matrix operand order checked by reading, not run on hardware
//
#include <hip/hip_runtime.h>
#include <hip/hip_bf16.h>
#include <math.h>

// Problem constants (B,S,D,H,FF) = (16,1024,1024,8,4096), DH=128
#define SB   16
#define SSq  1024
#define SD   1024
#define SH   8
#define SFF  4096
#define SDH  128
#define MTOT (SB * SSq)          // 16384 rows of activations

typedef __attribute__((ext_vector_type(16))) __bf16 bf16x16;
typedef __attribute__((ext_vector_type(8)))  __bf16 bf16x8;
typedef __attribute__((ext_vector_type(8)))  float  f32x8;

// ---------------------------------------------------------------------------
// CDNA5 async global->LDS copies (ASYNCcnt path), guarded so compile is safe
// on toolchains that don't declare the builtins.
// Prototype (from hipcc diagnostic): arg0 = AS1 (global) v4i*, arg1 = AS3
// (LDS) v4i*, then imm offset, imm cpol.
// ---------------------------------------------------------------------------
#if defined(__HIP_DEVICE_COMPILE__) &&                                        \
    __has_builtin(__builtin_amdgcn_global_load_async_to_lds_b128) &&          \
    __has_builtin(__builtin_amdgcn_s_wait_asynccnt)
#define USE_ASYNC_LDS 1
typedef int v4i_async __attribute__((__vector_size__(16)));
#define AS1 __attribute__((address_space(1)))
#define AS3 __attribute__((address_space(3)))
#endif

// Copy 16 bytes (one lane) from global to LDS.
__device__ __forceinline__ void copy16_g2l(const void* gsrc, void* ldst) {
#if defined(USE_ASYNC_LDS)
    __builtin_amdgcn_global_load_async_to_lds_b128(
        (AS1 v4i_async*)(void*)gsrc, (AS3 v4i_async*)ldst,
        /*imm offset*/ 0, /*cpol*/ 0);
#else
    *(uint4*)ldst = *(const uint4*)gsrc;
#endif
}

__device__ __forceinline__ void async_wait_all() {
#if defined(USE_ASYNC_LDS)
    __builtin_amdgcn_s_wait_asynccnt(0);
#endif
}

// Build a CDNA5 16-bit A/B WMMA fragment: per lane, K chunks [8g,8g+8) and
// [16+8g,16+8g+8) of a 32-wide K step, each an aligned 16B LDS load.
__device__ __forceinline__ bf16x16 make_frag(const __bf16* p, int g) {
    bf16x8 lo = *(const bf16x8*)(p + 8 * g);
    bf16x8 hi = *(const bf16x8*)(p + 16 + 8 * g);
    union { bf16x16 v; bf16x8 h[2]; } u;
    u.h[0] = lo; u.h[1] = hi;
    return u.v;
}

// ---------------------------------------------------------------------------
// fp32 -> bf16 conversion (4 elements / thread)
// ---------------------------------------------------------------------------
__global__ void cvt_bf16(const float* __restrict__ in, __bf16* __restrict__ out) {
    size_t i = ((size_t)blockIdx.x * 256 + threadIdx.x) * 4;
    float4 v = *(const float4*)(in + i);
    out[i + 0] = (__bf16)v.x;
    out[i + 1] = (__bf16)v.y;
    out[i + 2] = (__bf16)v.z;
    out[i + 3] = (__bf16)v.w;
}

// ---------------------------------------------------------------------------
// Generic WMMA GEMM: C[M,N] = A[M,K] @ B[N,K]^T (+bias[N]), 128x128 tile,
// 256 threads (8 waves), double-buffered LDS fed by async copies, K-step 32.
// MODE 0: store fp32   MODE 1: store bf16   MODE 2: store bf16 + ReLU
// MODE 3: store bf16 as per-head-transposed V  (VT[bh][dh][s'])
// ---------------------------------------------------------------------------
template <int MODE>
__global__ __launch_bounds__(256)
void gemm128(const __bf16* __restrict__ A, const __bf16* __restrict__ Bw,
             const float* __restrict__ bias, void* __restrict__ Cout,
             int M, int N, int K) {
    __shared__ __align__(16) __bf16 As[2][128][40];  // +8 pad: conflict-free frag reads
    __shared__ __align__(16) __bf16 Bs[2][128][40];

    const int tid  = threadIdx.x;
    const int tM   = blockIdx.y * 128;
    const int tN   = blockIdx.x * 128;
    const int lrow = tid >> 1;       // 0..127 staging row
    const int lseg = tid & 1;        // 16-elem segment within 32-wide K step
    const __bf16* Ag = A  + (size_t)(tM + lrow) * K + lseg * 16;
    const __bf16* Bg = Bw + (size_t)(tN + lrow) * K + lseg * 16;

    const int w    = tid >> 5;
    const int wm   = w >> 2;         // 0..1  -> 64-row strip
    const int wn   = w & 3;          // 0..3  -> 32-col strip
    const int lane = tid & 31;
    const int g    = lane >> 4;
    const int ml   = lane & 15;

    f32x8 acc[4][2] = {};
    const int nk = K >> 5;

    auto stage = [&](int b, int kb) {
        const __bf16* ap = Ag + (size_t)kb * 32;
        const __bf16* bp = Bg + (size_t)kb * 32;
        copy16_g2l(ap + 0, &As[b][lrow][lseg * 16 + 0]);
        copy16_g2l(ap + 8, &As[b][lrow][lseg * 16 + 8]);
        copy16_g2l(bp + 0, &Bs[b][lrow][lseg * 16 + 0]);
        copy16_g2l(bp + 8, &Bs[b][lrow][lseg * 16 + 8]);
    };

    stage(0, 0);
    async_wait_all();
    __syncthreads();

    int buf = 0;
    for (int kb = 0; kb < nk; ++kb) {
        if (kb + 1 < nk) {
            stage(buf ^ 1, kb + 1);   // async engine fills next buffer
            if (kb + 2 < nk) {        // gfx1250 global_prefetch_b8
                __builtin_prefetch(Ag + (size_t)(kb + 2) * 32, 0, 1);
                __builtin_prefetch(Bg + (size_t)(kb + 2) * 32, 0, 1);
            }
        }
        bf16x16 af[4], bfv[2];
#pragma unroll
        for (int i = 0; i < 4; ++i) af[i]  = make_frag(&As[buf][wm * 64 + i * 16 + ml][0], g);
#pragma unroll
        for (int j = 0; j < 2; ++j) bfv[j] = make_frag(&Bs[buf][wn * 32 + j * 16 + ml][0], g);
#pragma unroll
        for (int i = 0; i < 4; ++i)
#pragma unroll
            for (int j = 0; j < 2; ++j)
                acc[i][j] = __builtin_amdgcn_wmma_f32_16x16x32_bf16(
                    false, af[i], false, bfv[j], (short)0, acc[i][j], false, false);
        if (kb + 1 < nk) {
            async_wait_all();         // my async loads landed
            __syncthreads();          // everyone's landed + done reading `buf`
            buf ^= 1;
        }
    }

#pragma unroll
    for (int i = 0; i < 4; ++i)
#pragma unroll
        for (int j = 0; j < 2; ++j)
#pragma unroll
            for (int r = 0; r < 8; ++r) {
                int m = tM + wm * 64 + i * 16 + g * 8 + r;
                int n = tN + wn * 32 + j * 16 + ml;
                float v = acc[i][j][r];
                if (bias) v += bias[n];
                if constexpr (MODE == 2) v = fmaxf(v, 0.0f);
                if constexpr (MODE == 0) {
                    ((float*)Cout)[(size_t)m * N + n] = v;
                } else if constexpr (MODE == 1 || MODE == 2) {
                    ((__bf16*)Cout)[(size_t)m * N + n] = (__bf16)v;
                } else {  // MODE 3: scrambled-head transpose of V
                    int b  = m >> 10;
                    int bh = b * 8 + ((m & 1023) >> 7);
                    int sp = ((m & 127) << 3) + (n >> 7);
                    int dh = n & 127;
                    ((__bf16*)Cout)[((size_t)bh * SDH + dh) * SSq + sp] = (__bf16)v;
                }
            }
}

// ---------------------------------------------------------------------------
// Flash attention: per (bh, 128-query tile). 8 waves x 16 q-rows each.
// K-blocks of 64 keys async-staged in LDS. ctx written back to [B,S,D].
// ---------------------------------------------------------------------------
__global__ __launch_bounds__(256)
void flash_attn(const __bf16* __restrict__ Qh, const __bf16* __restrict__ Kh,
                const __bf16* __restrict__ VT, const float* __restrict__ bias,
                __bf16* __restrict__ Ctx) {
    __shared__ __align__(16) __bf16 Ks[64][136];    // [k-row][dh]
    __shared__ __align__(16) __bf16 Vs[128][72];    // [dh][k]
    __shared__ __align__(16) __bf16 Ps[8][16][72];  // per-wave P re-layout

    const int tid = threadIdx.x;
    const int bh  = blockIdx.y;
    const int q0  = blockIdx.x * 128;
    const int w   = tid >> 5;
    const int lane = tid & 31;
    const int g   = lane >> 4;
    const int ml  = lane & 15;
    const float scale = 0.08838834764831845f;  // 1/sqrt(128)

    bf16x16 qf[4];
    {
        const __bf16* qp = Qh + ((size_t)bh * SSq + (q0 + w * 16 + ml)) * SDH;
#pragma unroll
        for (int kk = 0; kk < 4; ++kk) qf[kk] = make_frag(qp + kk * 32, g);
    }

    f32x8 cacc[8] = {};
    float mrun[8], lrun[8];
#pragma unroll
    for (int r = 0; r < 8; ++r) { mrun[r] = -1e30f; lrun[r] = 0.0f; }

    const int krow = tid >> 2, kseg = tid & 3;
    const int vrow = tid >> 1, vseg = tid & 1;

    for (int kb = 0; kb < 16; ++kb) {
        const int k0 = kb * 64;
        {
            const __bf16* ksrc = Kh + ((size_t)bh * SSq + (k0 + krow)) * SDH + kseg * 32;
            const __bf16* vsrc = VT + ((size_t)bh * SDH + vrow) * SSq + k0 + vseg * 32;
#pragma unroll
            for (int u = 0; u < 4; ++u) {
                copy16_g2l(ksrc + u * 8, &Ks[krow][kseg * 32 + u * 8]);
                copy16_g2l(vsrc + u * 8, &Vs[vrow][vseg * 32 + u * 8]);
            }
        }
        async_wait_all();
        __syncthreads();

        // scores 16x64 for this wave
        f32x8 sacc[4] = {};
#pragma unroll
        for (int kk = 0; kk < 4; ++kk)
#pragma unroll
            for (int j = 0; j < 4; ++j) {
                bf16x16 bk = make_frag(&Ks[j * 16 + ml][kk * 32], g);
                sacc[j] = __builtin_amdgcn_wmma_f32_16x16x32_bf16(
                    false, qf[kk], false, bk, (short)0, sacc[j], false, false);
            }

        // bias + scale, block row-max
        float bmax[8];
#pragma unroll
        for (int r = 0; r < 8; ++r) bmax[r] = -1e30f;
#pragma unroll
        for (int j = 0; j < 4; ++j)
#pragma unroll
            for (int r = 0; r < 8; ++r) {
                int qg = q0 + w * 16 + g * 8 + r;
                int kg = k0 + j * 16 + ml;
                float s = (sacc[j][r] + bias[(size_t)qg * SSq + kg]) * scale;
                sacc[j][r] = s;
                bmax[r] = fmaxf(bmax[r], s);
            }
#pragma unroll
        for (int r = 0; r < 8; ++r)
#pragma unroll
            for (int off = 1; off < 16; off <<= 1)
                bmax[r] = fmaxf(bmax[r], __shfl_xor(bmax[r], off));

        float alpha[8], rsum[8];
#pragma unroll
        for (int r = 0; r < 8; ++r) {
            float mn = fmaxf(mrun[r], bmax[r]);
            alpha[r] = __expf(mrun[r] - mn);
            mrun[r]  = mn;
            rsum[r]  = 0.0f;
        }
#pragma unroll
        for (int j = 0; j < 4; ++j)
#pragma unroll
            for (int r = 0; r < 8; ++r) {
                float p = __expf(sacc[j][r] - mrun[r]);
                rsum[r] += p;
                Ps[w][g * 8 + r][j * 16 + ml] = (__bf16)p;
            }
#pragma unroll
        for (int r = 0; r < 8; ++r) {
#pragma unroll
            for (int off = 1; off < 16; off <<= 1) rsum[r] += __shfl_xor(rsum[r], off);
            lrun[r] = lrun[r] * alpha[r] + rsum[r];
#pragma unroll
            for (int jd = 0; jd < 8; ++jd) cacc[jd][r] *= alpha[r];
        }

        // ctx += P @ V  (wave-local LDS round-trip re-layouts C-frag -> A-frag)
#pragma unroll
        for (int ks = 0; ks < 2; ++ks) {
            bf16x16 pa = make_frag(&Ps[w][ml][ks * 32], g);
#pragma unroll
            for (int jd = 0; jd < 8; ++jd) {
                bf16x16 vb = make_frag(&Vs[jd * 16 + ml][ks * 32], g);
                cacc[jd] = __builtin_amdgcn_wmma_f32_16x16x32_bf16(
                    false, pa, false, vb, (short)0, cacc[jd], false, false);
            }
        }
        __syncthreads();
    }

    // normalize + un-scramble back to [B,S,D]
#pragma unroll
    for (int jd = 0; jd < 8; ++jd)
#pragma unroll
        for (int r = 0; r < 8; ++r) {
            int sp = q0 + w * 16 + g * 8 + r;
            int dh = jd * 16 + ml;
            float v = cacc[jd][r] / lrun[r];
            int b = bh >> 3, h = bh & 7;
            int s = h * 128 + (sp >> 3);
            int d = ((sp & 7) << 7) + dh;
            Ctx[((size_t)b * SSq + s) * SD + d] = (__bf16)v;
        }
}

// ---------------------------------------------------------------------------
// Residual + LayerNorm, fused fp32/bf16 dual output. One row per block.
// ---------------------------------------------------------------------------
__global__ __launch_bounds__(256)
void ln_kernel(const float* __restrict__ a, const float* __restrict__ b,
               const float* __restrict__ gamma, const float* __restrict__ beta,
               float* __restrict__ outf, __bf16* __restrict__ outb) {
    const int row = blockIdx.x;
    const float* pa = a + (size_t)row * SD;
    const float* pb = b + (size_t)row * SD;
    float s = 0.0f, ss = 0.0f, vals[4];
#pragma unroll
    for (int i = 0; i < 4; ++i) {
        int c = threadIdx.x + i * 256;
        float v = pa[c] + pb[c];
        vals[i] = v; s += v; ss += v * v;
    }
    __shared__ float red[2][8];
#pragma unroll
    for (int off = 16; off > 0; off >>= 1) {
        s  += __shfl_down(s, off);
        ss += __shfl_down(ss, off);
    }
    if ((threadIdx.x & 31) == 0) { red[0][threadIdx.x >> 5] = s; red[1][threadIdx.x >> 5] = ss; }
    __syncthreads();
    if (threadIdx.x == 0) {
        float S = 0, SS2 = 0;
        for (int i = 0; i < 8; ++i) { S += red[0][i]; SS2 += red[1][i]; }
        red[0][0] = S / SD; red[1][0] = SS2 / SD;
    }
    __syncthreads();
    float mu  = red[0][0];
    float inv = rsqrtf(red[1][0] - mu * mu + 1e-5f);
#pragma unroll
    for (int i = 0; i < 4; ++i) {
        int c = threadIdx.x + i * 256;
        float y = (vals[i] - mu) * inv * gamma[c] + beta[c];
        outf[(size_t)row * SD + c] = y;
        if (outb) outb[(size_t)row * SD + c] = (__bf16)y;
    }
}

// ---------------------------------------------------------------------------
// attn[0] recompute: softmax probabilities for head-batch 0 (tiny).
// ---------------------------------------------------------------------------
__global__ __launch_bounds__(256)
void attn0_kernel(const __bf16* __restrict__ Qh, const __bf16* __restrict__ Kh,
                  const float* __restrict__ bias, float* __restrict__ outp) {
    const int q = blockIdx.x;
    __shared__ float qrow[128];
    __shared__ float sc[1024];
    __shared__ float red[8], red2[8];
    if (threadIdx.x < 128) qrow[threadIdx.x] = (float)Qh[(size_t)q * SDH + threadIdx.x];
    __syncthreads();
    float lmax = -1e30f;
    const float scale = 0.08838834764831845f;
    for (int t = 0; t < 4; ++t) {
        int k = threadIdx.x + t * 256;
        const __bf16* kr = Kh + (size_t)k * SDH;
        float d = 0.0f;
        for (int i = 0; i < 128; ++i) d += qrow[i] * (float)kr[i];
        float sv = (d + bias[(size_t)q * SSq + k]) * scale;
        sc[k] = sv; lmax = fmaxf(lmax, sv);
    }
    for (int off = 16; off > 0; off >>= 1) lmax = fmaxf(lmax, __shfl_xor(lmax, off));
    if ((threadIdx.x & 31) == 0) red[threadIdx.x >> 5] = lmax;
    __syncthreads();
    if (threadIdx.x == 0) {
        float m = red[0];
        for (int i = 1; i < 8; ++i) m = fmaxf(m, red[i]);
        red[0] = m;
    }
    __syncthreads();
    float m = red[0], lsum = 0.0f;
    for (int t = 0; t < 4; ++t) {
        int k = threadIdx.x + t * 256;
        float e = __expf(sc[k] - m);
        sc[k] = e; lsum += e;
    }
    for (int off = 16; off > 0; off >>= 1) lsum += __shfl_xor(lsum, off);
    if ((threadIdx.x & 31) == 0) red2[threadIdx.x >> 5] = lsum;
    __syncthreads();
    if (threadIdx.x == 0) {
        float s2 = 0;
        for (int i = 0; i < 8; ++i) s2 += red2[i];
        red2[0] = s2;
    }
    __syncthreads();
    float inv = 1.0f / red2[0];
    for (int t = 0; t < 4; ++t) {
        int k = threadIdx.x + t * 256;
        outp[(size_t)q * SSq + k] = sc[k] * inv;
    }
}

// ---------------------------------------------------------------------------
// Launcher (needs ws_size >= ~540 MB)
// ---------------------------------------------------------------------------
extern "C" void kernel_launch(void* const* d_in, const int* in_sizes, int n_in,
                              void* d_out, int out_size, void* d_ws, size_t ws_size,
                              hipStream_t stream) {
    (void)in_sizes; (void)n_in; (void)out_size; (void)ws_size;
    const float* x    = (const float*)d_in[0];
    const float* bias = (const float*)d_in[1];
    const float* Wq   = (const float*)d_in[2];
    const float* bq   = (const float*)d_in[3];
    const float* Wk   = (const float*)d_in[4];
    const float* bk   = (const float*)d_in[5];
    const float* Wv   = (const float*)d_in[6];
    const float* bv   = (const float*)d_in[7];
    const float* Wo   = (const float*)d_in[8];
    const float* bo   = (const float*)d_in[9];
    const float* g1   = (const float*)d_in[10];
    const float* be1  = (const float*)d_in[11];
    const float* W1   = (const float*)d_in[12];
    const float* b1   = (const float*)d_in[13];
    const float* W2   = (const float*)d_in[14];
    const float* b2   = (const float*)d_in[15];
    const float* g2   = (const float*)d_in[16];
    const float* be2  = (const float*)d_in[17];

    char* ws = (char*)d_ws;
    const size_t MB = 1024u * 1024u;
    __bf16* Xb    = (__bf16*)(ws + 0);          // 32 MB
    __bf16* Wqb   = (__bf16*)(ws + 32 * MB);    //  2 MB
    __bf16* Wkb   = (__bf16*)(ws + 34 * MB);
    __bf16* Wvb   = (__bf16*)(ws + 36 * MB);
    __bf16* Wob   = (__bf16*)(ws + 38 * MB);
    __bf16* W1b   = (__bf16*)(ws + 40 * MB);    //  8 MB
    __bf16* W2b   = (__bf16*)(ws + 48 * MB);    //  8 MB
    __bf16* Qh    = (__bf16*)(ws + 56 * MB);    // 32 MB  [bh][s'][dh]
    __bf16* Kh    = (__bf16*)(ws + 88 * MB);    // 32 MB
    __bf16* VT    = (__bf16*)(ws + 120 * MB);   // 32 MB  [bh][dh][s']
    __bf16* Ctxb  = (__bf16*)(ws + 152 * MB);   // 32 MB  [B,S,D]
    float*  attnf = (float*) (ws + 184 * MB);   // 64 MB
    float*  out1f = (float*) (ws + 248 * MB);   // 64 MB
    __bf16* out1b = (__bf16*)(ws + 312 * MB);   // 32 MB
    __bf16* hb    = (__bf16*)(ws + 344 * MB);   // 128 MB
    float*  fff   = (float*) (ws + 472 * MB);   // 64 MB

    const dim3 blk(256);
    // fp32 -> bf16 conversions (4 elems/thread)
    cvt_bf16<<<16384, blk, 0, stream>>>(x,  Xb);
    cvt_bf16<<<1024,  blk, 0, stream>>>(Wq, Wqb);
    cvt_bf16<<<1024,  blk, 0, stream>>>(Wk, Wkb);
    cvt_bf16<<<1024,  blk, 0, stream>>>(Wv, Wvb);
    cvt_bf16<<<1024,  blk, 0, stream>>>(Wo, Wob);
    cvt_bf16<<<4096,  blk, 0, stream>>>(W1, W1b);
    cvt_bf16<<<4096,  blk, 0, stream>>>(W2, W2b);

    // QKV projections (Q,K flat == scrambled head layout; V stored transposed)
    gemm128<1><<<dim3(8, 128),  blk, 0, stream>>>(Xb, Wqb, bq, Qh,   MTOT, SD,  SD);
    gemm128<1><<<dim3(8, 128),  blk, 0, stream>>>(Xb, Wkb, bk, Kh,   MTOT, SD,  SD);
    gemm128<3><<<dim3(8, 128),  blk, 0, stream>>>(Xb, Wvb, bv, VT,   MTOT, SD,  SD);

    // attention
    flash_attn<<<dim3(8, 128), blk, 0, stream>>>(Qh, Kh, VT, bias, Ctxb);
    attn0_kernel<<<1024, blk, 0, stream>>>(Qh, Kh, bias, (float*)d_out + (size_t)SB * SSq * SD);

    // output projection + LN1
    gemm128<0><<<dim3(8, 128),  blk, 0, stream>>>(Ctxb, Wob, bo, attnf, MTOT, SD,  SD);
    ln_kernel<<<MTOT, blk, 0, stream>>>(attnf, x, g1, be1, out1f, out1b);

    // FFN + LN2
    gemm128<2><<<dim3(32, 128), blk, 0, stream>>>(out1b, W1b, b1, hb,  MTOT, SFF, SD);
    gemm128<0><<<dim3(8, 128),  blk, 0, stream>>>(hb,    W2b, b2, fff, MTOT, SD,  SFF);
    ln_kernel<<<MTOT, blk, 0, stream>>>(fff, out1f, g2, be2, (float*)d_out, nullptr);
}
